// Simple_Model_62998580297759
// MI455X (gfx1250) — compile-verified, matrix operand order checked by reading, not run on hardware
//
#include <hip/hip_runtime.h>
#include <hip/hip_bf16.h>
#include <math.h>

// ---------------- constants (from reference) ----------------
#define FD    178              // node feature dim
#define KP    192              // padded K / activation stride (6 * 32)
#define KT    6                // k-steps of 32
#define NT    12               // output col tiles of 16 (192 padded)
#define WFRAG_ELEMS (NT * KT * 32 * 16)   // 36864 bf16 per packed matrix

typedef __attribute__((ext_vector_type(16))) __bf16         v16bf;
typedef __attribute__((ext_vector_type(16))) unsigned short v16u;
typedef __attribute__((ext_vector_type(8)))  float          v8f;
typedef __attribute__((ext_vector_type(4)))  unsigned       v4u;
typedef __attribute__((ext_vector_type(8)))  int            v8i;
typedef __attribute__((ext_vector_type(4)))  int            v4i;

__device__ __forceinline__ unsigned short f2bf(float f) {
  unsigned u = __float_as_uint(f);
  u += 0x7fffu + ((u >> 16) & 1u);          // round to nearest even
  return (unsigned short)(u >> 16);
}
__device__ __forceinline__ float bf2f(unsigned short u) {
  return __uint_as_float((unsigned)u << 16);
}

// order-preserving float <-> uint map for atomicMax-based segment_max
__device__ __forceinline__ unsigned fenc(float f) {
  unsigned u = __float_as_uint(f);
  return (u & 0x80000000u) ? ~u : (u | 0x80000000u);
}
__device__ __forceinline__ float fdec(unsigned u) {
  unsigned b = (u & 0x80000000u) ? (u & 0x7fffffffu) : ~u;
  return __uint_as_float(b);
}

// ---------------- weight repack: [178][178] f32 -> WMMA B-fragment bf16 ----------------
__global__ void pack_w_kernel(const float* __restrict__ W, unsigned short* __restrict__ out) {
  int t = blockIdx.x * blockDim.x + threadIdx.x;
  if (t >= WFRAG_ELEMS) return;
  int i    = t & 15;
  int lane = (t >> 4) & 31;
  int kk   = (t >> 9) % KT;
  int ct   = (t >> 9) / KT;
  // B fragment (16-bit, 32x16): lane -> N = ct*16 + lane%16,
  // element i -> K = kk*32 + (lane>=16 ? 8 : 0) + (i&7) + (i>>3)*16
  int K  = kk * 32 + ((lane >> 4) << 3) + (i & 7) + ((i >> 3) << 4);
  int Nn = ct * 16 + (lane & 15);
  float v = (K < FD && Nn < FD) ? W[K * FD + Nn] : 0.0f;
  out[t] = f2bf(v);
}

// ---------------- activation convert: f32 [N,178] -> bf16 [N,192] (zero pad) ------------
__global__ void to_bf16_pad_kernel(const float* __restrict__ in, unsigned short* __restrict__ out,
                                   int nrows, int doRelu) {
  int t = blockIdx.x * blockDim.x + threadIdx.x;
  if (t >= nrows * KP) return;
  int r = t / KP, c = t % KP;
  float v = 0.f;
  if (c < FD) {
    v = in[(size_t)r * FD + c];
    if (doRelu) v = fmaxf(v, 0.f);
  }
  out[t] = f2bf(v);
}

// ---------------- fused WMMA GEMM: {Q,K,V,S}[N,178] = Xb[N,192] @ W{q,k,v,s} + b --------
__global__ void __launch_bounds__(128)
gemm_qkvs_wmma_kernel(const unsigned short* __restrict__ Xb,      // [nrows, 192] bf16
                      const unsigned short* __restrict__ Wfrag4,  // 4 packed matrices
                      const float* __restrict__ b0, const float* __restrict__ b1,
                      const float* __restrict__ b2, const float* __restrict__ b3,
                      float* __restrict__ Y0, float* __restrict__ Y1,
                      float* __restrict__ Y2, float* __restrict__ Y3,
                      int nrows) {
  __shared__ unsigned short tile[4][16][KP];   // 24 KB: 4 waves x 16 rows x 192 bf16
  const int wave = __builtin_amdgcn_readfirstlane((int)(threadIdx.x >> 5));
  const int lane = threadIdx.x & 31;
  const int rowbase = blockIdx.x * 64 + wave * 16;

#if __has_builtin(__builtin_amdgcn_tensor_load_to_lds) && __has_builtin(__builtin_amdgcn_s_wait_tensorcnt)
  {
    // Tensor Data Mover: stage this wave's 16x192 bf16 tile into LDS.
    int remRows = nrows - rowbase;
    if (remRows < 0) remRows = 0;
    unsigned long long ga = (unsigned long long)(uintptr_t)(Xb + (size_t)rowbase * KP);
    unsigned ldsOff = (unsigned)(uintptr_t)(&tile[wave][0][0]);   // flat LDS addr low 32b == LDS offset
    v4u g0;
    g0[0] = 1u;                                        // count=1, user descriptor
    g0[1] = ldsOff;                                    // lds_addr
    g0[2] = (unsigned)(ga & 0xffffffffu);              // global_addr[31:0]
    g0[3] = (unsigned)((ga >> 32) & 0x1ffffffu) | (2u << 30);  // global_addr[56:32] | type=2
    v8i g1;
    g1[0] = (int)(1u << 16);                           // data_size=1 (2 bytes), mask=0
    g1[1] = (int)((unsigned)KP << 16);                 // tensor_dim0[15:0]=192
    g1[2] = (int)(((unsigned)remRows & 0xffffu) << 16);// tensor_dim0 hi=0 | tensor_dim1 lo
    g1[3] = (int)((((unsigned)remRows >> 16) & 0xffffu) | ((unsigned)KP << 16)); // dim1 hi | tile_dim0=192
    g1[4] = 16;                                        // tile_dim1=16, tile_dim2=0
    g1[5] = KP;                                        // tensor_dim0_stride=192
    g1[6] = 0;
    g1[7] = 0;
    v4i gz  = {0, 0, 0, 0};
    v8i gz8 = {0, 0, 0, 0, 0, 0, 0, 0};
    __builtin_amdgcn_tensor_load_to_lds(g0, g1, gz, gz, gz8, 0);
    __builtin_amdgcn_s_wait_tensorcnt(0);
    __asm__ volatile("" ::: "memory");
  }
#else
  for (int idx = lane; idx < 16 * KP; idx += 32) {
    int r = idx / KP, c = idx % KP;
    int gr = rowbase + r;
    tile[wave][r][c] = (gr < nrows) ? Xb[(size_t)gr * KP + c] : (unsigned short)0;
  }
  __asm__ volatile("" ::: "memory");
#endif

  const int m     = lane & 15;
  const int khalf = (lane >> 4) << 3;   // 0 or 8
  const float* biases[4] = {b0, b1, b2, b3};
  float*       outs[4]   = {Y0, Y1, Y2, Y3};
  const v8f vzero = {0.f, 0.f, 0.f, 0.f, 0.f, 0.f, 0.f, 0.f};

  for (int mat = 0; mat < 4; ++mat) {
    const unsigned short* Wf = Wfrag4 + (size_t)mat * WFRAG_ELEMS;
    v8f acc[NT];
    #pragma unroll
    for (int ct = 0; ct < NT; ++ct) acc[ct] = vzero;

    for (int kk = 0; kk < KT; ++kk) {
      // A fragment (16-bit, 16x32): element i -> K = kk*32 + khalf + (i&7) + (i>>3)*16
      v16u au;
      const int kbase = kk * 32 + khalf;
      #pragma unroll
      for (int i = 0; i < 16; ++i)
        au[i] = tile[wave][m][kbase + (i & 7) + ((i >> 3) << 4)];
      v16bf a = __builtin_bit_cast(v16bf, au);

      #pragma unroll
      for (int ct = 0; ct < NT; ++ct) {
        v16u bu = *reinterpret_cast<const v16u*>(Wf + ((size_t)((ct * KT + kk) * 32 + lane) << 4));
        v16bf b = __builtin_bit_cast(v16bf, bu);
        acc[ct] = __builtin_amdgcn_wmma_f32_16x16x32_bf16(
            false, a, false, b, (short)0, acc[ct], false, false);
      }
    }

    // C/D layout: VGPR r -> M = r + khalf ; lane -> N = ct*16 + lane%16
    const float* bias = biases[mat];
    float*       Y    = outs[mat];
    #pragma unroll
    for (int ct = 0; ct < NT; ++ct) {
      int n = ct * 16 + (lane & 15);
      if (n >= FD) continue;
      float bv = bias[n];
      #pragma unroll
      for (int r = 0; r < 8; ++r) {
        int gr = rowbase + r + khalf;
        if (gr < nrows) Y[(size_t)gr * FD + n] = acc[ct][r] + bv;
      }
    }
  }
}

// ---------------- edge kernels (wave32 per edge) ----------------
__global__ void edge_logits_kernel(const float* __restrict__ q, const float* __restrict__ k,
                                   const int* __restrict__ src, const int* __restrict__ dst,
                                   float* __restrict__ logits, int E) {
  int gid = blockIdx.x * blockDim.x + threadIdx.x;
  int e = gid >> 5, lane = gid & 31;
  if (e >= E) return;
  const float* qp = q + (size_t)dst[e] * FD;
  const float* kp = k + (size_t)src[e] * FD;
  float acc = 0.f;
  for (int c = lane; c < FD; c += 32) acc += qp[c] * kp[c];
  #pragma unroll
  for (int off = 16; off; off >>= 1) acc += __shfl_xor(acc, off, 32);
  if (lane == 0) logits[e] = acc * 0.07495316889958614f;   // 1/sqrt(178)
}

__global__ void edge_max_kernel(const float* __restrict__ logits, const int* __restrict__ dst,
                                unsigned* __restrict__ mEnc, int E) {
  int e = blockIdx.x * blockDim.x + threadIdx.x;
  if (e >= E) return;
  atomicMax(&mEnc[dst[e]], fenc(logits[e]));
}

__global__ void edge_expsum_kernel(const float* __restrict__ logits, const int* __restrict__ dst,
                                   const unsigned* __restrict__ mEnc, float* __restrict__ ex,
                                   float* __restrict__ denom, int E) {
  int e = blockIdx.x * blockDim.x + threadIdx.x;
  if (e >= E) return;
  int d = dst[e];
  float v = __expf(logits[e] - fdec(mEnc[d]));
  ex[e] = v;
  atomicAdd(&denom[d], v);
}

__global__ void edge_aggregate_kernel(const float* __restrict__ ex, const float* __restrict__ denom,
                                      const float* __restrict__ vmat,
                                      const int* __restrict__ src, const int* __restrict__ dst,
                                      float* __restrict__ accum, int E) {
  int gid = blockIdx.x * blockDim.x + threadIdx.x;
  int e = gid >> 5, lane = gid & 31;
  if (e >= E) return;
  int s = src[e], d = dst[e];
  float alpha = ex[e] / denom[d];
  const float* vp = vmat + (size_t)s * FD;
  float* ap = accum + (size_t)d * FD;
  for (int c = lane; c < FD; c += 32) atomicAdd(&ap[c], alpha * vp[c]);
}

// ---------------- fill / pool / dense / concat ----------------
__global__ void fill_f32_kernel(float* p, float v, int n) {
  int i = blockIdx.x * blockDim.x + threadIdx.x;
  if (i < n) p[i] = v;
}
__global__ void fill_u32_kernel(unsigned* p, unsigned v, int n) {
  int i = blockIdx.x * blockDim.x + threadIdx.x;
  if (i < n) p[i] = v;
}

__global__ void pool_kernel(const unsigned short* __restrict__ hb, const int* __restrict__ batch,
                            float* __restrict__ sums, float* __restrict__ cnt, int nnodes) {
  int gid = blockIdx.x * blockDim.x + threadIdx.x;
  int node = gid >> 5, lane = gid & 31;
  if (node >= nnodes) return;
  int b = batch[node];
  const unsigned short* hp = hb + (size_t)node * KP;
  float* sp = sums + (size_t)b * FD;
  for (int c = lane; c < FD; c += 32) atomicAdd(&sp[c], bf2f(hp[c]));
  if (lane == 0) atomicAdd(&cnt[b], 1.0f);
}

__global__ void dense_kernel(const float* __restrict__ A, const float* __restrict__ W,
                             const float* __restrict__ bias, float* __restrict__ Y,
                             int rows, int K, int Nout, int doRelu) {
  int t = blockIdx.x * blockDim.x + threadIdx.x;
  if (t >= rows * Nout) return;
  int r = t / Nout, n = t % Nout;
  float acc = bias[n];
  const float* ap = A + (size_t)r * K;
  for (int k = 0; k < K; ++k) acc += ap[k] * W[(size_t)k * Nout + n];
  if (doRelu) acc = fmaxf(acc, 0.f);
  Y[t] = acc;
}

__global__ void concat_kernel(const float* __restrict__ sums, const float* __restrict__ cnt,
                              const float* __restrict__ g2, float* __restrict__ z, int rows) {
  const int ZD = FD + 64;
  int t = blockIdx.x * blockDim.x + threadIdx.x;
  if (t >= rows * ZD) return;
  int r = t / ZD, c = t % ZD;
  z[t] = (c < FD) ? sums[(size_t)r * FD + c] / fmaxf(cnt[r], 1.f)
                  : g2[(size_t)r * 64 + (c - FD)];
}

// ---------------- host orchestration ----------------
extern "C" void kernel_launch(void* const* d_in, const int* in_sizes, int n_in,
                              void* d_out, int out_size, void* d_ws, size_t ws_size,
                              hipStream_t stream) {
  (void)n_in; (void)ws_size;
  const float* x     = (const float*)d_in[0];
  const int*   ei    = (const int*)d_in[1];
  const float* gf    = (const float*)d_in[2];
  const int*   batch = (const int*)d_in[3];

  const int N  = in_sizes[3];
  const int E  = in_sizes[1] / 2;
  const int G  = out_size;
  const int GFdim = in_sizes[2] / G;
  const int* src = ei;
  const int* dst = ei + E;

  // workspace bump allocator
  char* ws = (char*)d_ws;
  size_t off = 0;
  auto walloc = [&](size_t bytes) -> void* {
    off = (off + 255) & ~(size_t)255;
    void* p = ws + off;
    off += bytes;
    return p;
  };
  unsigned short* wfrag = (unsigned short*)walloc((size_t)12 * WFRAG_ELEMS * sizeof(unsigned short));
  unsigned short* Hb  = (unsigned short*)walloc((size_t)N * KP * 2);  // bf16 activations [N,192]
  float* Q    = (float*)walloc((size_t)N * FD * 4);
  float* Kb   = (float*)walloc((size_t)N * FD * 4);
  float* V    = (float*)walloc((size_t)N * FD * 4);
  float* S    = (float*)walloc((size_t)N * FD * 4);   // skip output, doubles as aggregator
  float* logits = (float*)walloc((size_t)E * 4);
  float* ex     = (float*)walloc((size_t)E * 4);
  unsigned* mEnc = (unsigned*)walloc((size_t)N * 4);
  float* denom  = (float*)walloc((size_t)N * 4);
  float* pool   = (float*)walloc((size_t)G * FD * 4);
  float* cnt    = (float*)walloc((size_t)G * 4);
  float* g1     = (float*)walloc((size_t)G * 64 * 4);
  float* g2     = (float*)walloc((size_t)G * 64 * 4);
  float* z      = (float*)walloc((size_t)G * (FD + 64) * 4);
  float* m1     = (float*)walloc((size_t)G * 512 * 4);
  float* m2     = (float*)walloc((size_t)G * 512 * 4);
  float* m3     = (float*)walloc((size_t)G * 128 * 4);

  // pack all 12 conv weight matrices into WMMA fragment order
  const int packBlocks = (WFRAG_ELEMS + 255) / 256;
  for (int l = 0; l < 3; ++l)
    for (int m = 0; m < 4; ++m)
      pack_w_kernel<<<packBlocks, 256, 0, stream>>>(
          (const float*)d_in[4 + l * 8 + m * 2], wfrag + (size_t)(l * 4 + m) * WFRAG_ELEMS);

  const int gemmBlocks  = (N + 63) / 64;
  const int edgeWBlocks = (E * 32 + 255) / 256;
  const int edgeBlocks  = (E + 255) / 256;
  const int actBlocks   = (N * KP + 255) / 256;

  // layer-0 input: convert x to padded bf16
  to_bf16_pad_kernel<<<actBlocks, 256, 0, stream>>>(x, Hb, N, 0);

  for (int l = 0; l < 3; ++l) {
    const unsigned short* wl = wfrag + (size_t)(l * 4) * WFRAG_ELEMS;
    const float* qb = (const float*)d_in[4 + l * 8 + 1];
    const float* kb = (const float*)d_in[4 + l * 8 + 3];
    const float* vb = (const float*)d_in[4 + l * 8 + 5];
    const float* sb = (const float*)d_in[4 + l * 8 + 7];

    gemm_qkvs_wmma_kernel<<<gemmBlocks, 128, 0, stream>>>(
        Hb, wl, qb, kb, vb, sb, Q, Kb, V, S, N);

    edge_logits_kernel<<<edgeWBlocks, 256, 0, stream>>>(Q, Kb, src, dst, logits, E);
    fill_u32_kernel<<<(N + 255) / 256, 256, 0, stream>>>(mEnc, 0u, N);
    fill_f32_kernel<<<(N + 255) / 256, 256, 0, stream>>>(denom, 0.f, N);
    edge_max_kernel<<<edgeBlocks, 256, 0, stream>>>(logits, dst, mEnc, E);
    edge_expsum_kernel<<<edgeBlocks, 256, 0, stream>>>(logits, dst, mEnc, ex, denom, E);
    edge_aggregate_kernel<<<edgeWBlocks, 256, 0, stream>>>(ex, denom, V, src, dst, S, E);
    // relu(agg + skip) -> next-layer bf16 activations (pad cols zeroed)
    to_bf16_pad_kernel<<<actBlocks, 256, 0, stream>>>(S, Hb, N, 1);
  }

  // global mean pool (reads bf16 activations)
  fill_f32_kernel<<<(G * FD + 255) / 256, 256, 0, stream>>>(pool, 0.f, G * FD);
  fill_f32_kernel<<<(G + 255) / 256, 256, 0, stream>>>(cnt, 0.f, G);
  pool_kernel<<<(N * 32 + 255) / 256, 256, 0, stream>>>(Hb, batch, pool, cnt, N);

  // MLP head
  auto launch_dense = [&](const float* A, int wi, float* Y, int rows, int K, int Nout, int relu) {
    int threads = rows * Nout;
    dense_kernel<<<(threads + 255) / 256, 256, 0, stream>>>(
        A, (const float*)d_in[wi], (const float*)d_in[wi + 1], Y, rows, K, Nout, relu);
  };
  launch_dense(gf, 28, g1, G, GFdim, 64, 1);
  launch_dense(g1, 30, g2, G, 64, 64, 1);
  concat_kernel<<<(G * (FD + 64) + 255) / 256, 256, 0, stream>>>(pool, cnt, g2, z, G);
  launch_dense(z,  32, m1, G, FD + 64, 512, 1);
  launch_dense(m1, 34, m2, G, 512, 512, 1);
  launch_dense(m2, 36, m3, G, 512, 128, 1);
  launch_dense(m3, 38, (float*)d_out, G, 128, 1, 0);
}
